// WanAttention_42640435314809
// MI455X (gfx1250) — matrix operand order repaired and checked by reading, not compiled, not run on hardware
//
#include <hip/hip_runtime.h>
#include <hip/hip_bf16.h>
#include <stdint.h>

// Dims fixed by the reference
#define B_  2
#define N_  2048
#define D_  2048
#define H_  16
#define E_  128
#define EPS_ 1e-5f

typedef __attribute__((ext_vector_type(16))) __bf16       bf16x16;
typedef __attribute__((ext_vector_type(8)))  float        f32x8;
typedef __attribute__((ext_vector_type(4)))  unsigned int u32x4;
typedef __attribute__((ext_vector_type(4)))  int          i32x4;

// CDNA5 async global->LDS copy (ASYNCcnt-tracked), if the toolchain has it.
#if defined(__gfx1250__) && \
    __has_builtin(__builtin_amdgcn_global_load_async_to_lds_b128) && \
    __has_builtin(__builtin_amdgcn_s_wait_asynccnt)
#define USE_ASYNC_LDS 1
#else
#define USE_ASYNC_LDS 0
#endif

#define AS1 __attribute__((address_space(1)))
#define AS3 __attribute__((address_space(3)))

union Frag16 {
  bf16x16 v;
  u32x4   q[2];
  unsigned short us[16];
};

__device__ __forceinline__ unsigned short f32_to_bf16(float f) {
  union { float f; unsigned int u; } x; x.f = f;
  unsigned int r = x.u + 0x7FFFu + ((x.u >> 16) & 1u);   // RNE
  return (unsigned short)(r >> 16);
}

__device__ __forceinline__ f32x8 wmma_bf16(const Frag16& a, const Frag16& b, f32x8 c) {
  // v_wmma_f32_16x16x32_bf16 : D = A(16x32) * B(32x16) + C
  return __builtin_amdgcn_wmma_f32_16x16x32_bf16(false, a.v, false, b.v,
                                                 (short)0, c, false, false);
}

// ---------------------------------------------------------------------------
// GEMM: C[M,Nc] = A[M,K] @ W[K,Nc], A/W fp32 in global, bf16 WMMA mainloop.
// mode 0: C = acc + bias                      (QKV projection)
// mode 1: C = residual + gate*(acc + bias)    (out projection)
// Block = 128 threads (4 waves). Tile 128x64, k-step 32.
// Each wave owns 32 rows (2 A-frags); B-frags shared across both -> 8 WMMA/step.
// ---------------------------------------------------------------------------
__global__ __launch_bounds__(128) void gemm_bf16_epilogue(
    const float* __restrict__ A, const float* __restrict__ W,
    const float* __restrict__ bias, float* __restrict__ C,
    const float* __restrict__ residual, const float* __restrict__ gate,
    int K, int Nc, int rows_per_b, int mode)
{
  __shared__ __align__(16) unsigned short As[128 * 32];  // [m][k] bf16
  __shared__ __align__(16) unsigned short Bs[64 * 32];   // [n][k] bf16 (transposed)

  const int m0 = blockIdx.y * 128;
  const int n0 = blockIdx.x * 64;
  const int t    = threadIdx.x;
  const int wave = t >> 5;
  const int lane = t & 31;
  const int lh   = lane >> 4;     // lane half (0/1)
  const int l15  = lane & 15;

  f32x8 acc[2][4];
#pragma unroll
  for (int af = 0; af < 2; ++af)
#pragma unroll
    for (int nt = 0; nt < 4; ++nt)
#pragma unroll
      for (int r = 0; r < 8; ++r) acc[af][nt][r] = 0.f;

  for (int k0 = 0; k0 < K; k0 += 32) {
    // --- A tile 128x32: float4 loads, packed bf16 pair stores ---
#pragma unroll
    for (int i = 0; i < 8; ++i) {
      int idx = t + i * 128;               // 1024 float4 slots
      int row = idx >> 3, c4 = idx & 7;
      const float4 v = *(const float4*)(A + (size_t)(m0 + row) * K + k0 + c4 * 4);
      unsigned int* d = (unsigned int*)(As + row * 32 + c4 * 4);
      d[0] = (unsigned)f32_to_bf16(v.x) | ((unsigned)f32_to_bf16(v.y) << 16);
      d[1] = (unsigned)f32_to_bf16(v.z) | ((unsigned)f32_to_bf16(v.w) << 16);
    }
    // --- W tile 32x64 -> transposed into Bs[n][k] ---
#pragma unroll
    for (int i = 0; i < 4; ++i) {
      int idx = t + i * 128;               // 512 float4 slots
      int kr = idx >> 4, c4 = idx & 15;
      const float4 v = *(const float4*)(W + (size_t)(k0 + kr) * Nc + n0 + c4 * 4);
      Bs[(c4 * 4 + 0) * 32 + kr] = f32_to_bf16(v.x);
      Bs[(c4 * 4 + 1) * 32 + kr] = f32_to_bf16(v.y);
      Bs[(c4 * 4 + 2) * 32 + kr] = f32_to_bf16(v.z);
      Bs[(c4 * 4 + 3) * 32 + kr] = f32_to_bf16(v.w);
    }
    // prefetch next k-tile into L2 while we compute (global_prefetch_b8)
    if (k0 + 32 < K) {
      __builtin_prefetch(A + (size_t)(m0 + (t >> 3) * 4) * K + k0 + 32, 0, 1);
      __builtin_prefetch(W + (size_t)(k0 + 32 + (t >> 4)) * Nc + n0, 0, 1);
    }
    __syncthreads();

    // A fragments: row = l15, K = (e<8?e:e+8) + lh*8  -> two b128 LDS loads
    Frag16 a[2];
#pragma unroll
    for (int af = 0; af < 2; ++af) {
      const u32x4* pa = (const u32x4*)(As + (wave * 32 + af * 16 + l15) * 32 + lh * 8);
      a[af].q[0] = pa[0];
      a[af].q[1] = pa[2];
    }
#pragma unroll
    for (int nt = 0; nt < 4; ++nt) {
      // B fragment: col = l15, K = (e<8?e:e+8) + lh*8 -> two b128 LDS loads
      Frag16 b;
      const u32x4* pb = (const u32x4*)(Bs + (nt * 16 + l15) * 32 + lh * 8);
      b.q[0] = pb[0]; b.q[1] = pb[2];
#pragma unroll
      for (int af = 0; af < 2; ++af)
        acc[af][nt] = wmma_bf16(a[af], b, acc[af][nt]);
    }
    __syncthreads();
  }

#pragma unroll
  for (int af = 0; af < 2; ++af) {
#pragma unroll
    for (int nt = 0; nt < 4; ++nt) {
#pragma unroll
      for (int r = 0; r < 8; ++r) {
        int gm = m0 + wave * 32 + af * 16 + lh * 8 + r;
        int gn = n0 + nt * 16 + l15;
        float v = acc[af][nt][r] + bias[gn];
        if (mode == 1) {
          int bb = gm / rows_per_b;
          v = residual[(size_t)gm * Nc + gn] + gate[(size_t)bb * Nc + gn] * v;
        }
        C[(size_t)gm * Nc + gn] = v;
      }
    }
  }
}

// ---------------------------------------------------------------------------
// Per-token: RMSNorm(q,k) over D, RoPE on interleaved pairs, split heads,
// pack q/k/v to bf16 in [B,H,N,E] layout. One 256-thread block per token.
// ---------------------------------------------------------------------------
__global__ __launch_bounds__(256) void norm_rope_pack(
    const float* __restrict__ qkv,
    const float* __restrict__ nqw, const float* __restrict__ nkw,
    const float* __restrict__ cosT, const float* __restrict__ sinT,
    unsigned short* __restrict__ qb, unsigned short* __restrict__ kb,
    unsigned short* __restrict__ vb)
{
  const int bn = blockIdx.x;
  const int b  = bn >> 11;        // N_ = 2048
  const int n  = bn & (N_ - 1);
  const float* row = qkv + (size_t)bn * (3 * D_);

  float sq = 0.f, sk = 0.f;
  for (int i = threadIdx.x; i < D_; i += 256) {
    float q = row[i];       sq += q * q;
    float k = row[D_ + i];  sk += k * k;
  }
#pragma unroll
  for (int o = 16; o > 0; o >>= 1) {
    sq += __shfl_xor(sq, o, 32);
    sk += __shfl_xor(sk, o, 32);
  }
  __shared__ float redq[8], redk[8];
  const int wave = threadIdx.x >> 5;
  if ((threadIdx.x & 31) == 0) { redq[wave] = sq; redk[wave] = sk; }
  __syncthreads();
  if (threadIdx.x == 0) {
    float a = 0.f, c = 0.f;
#pragma unroll
    for (int i = 0; i < 8; ++i) { a += redq[i]; c += redk[i]; }
    redq[0] = rsqrtf(a / (float)D_ + EPS_);
    redk[0] = rsqrtf(c / (float)D_ + EPS_);
  }
  __syncthreads();
  const float rq = redq[0], rk = redk[0];

  for (int p = threadIdx.x; p < D_ / 2; p += 256) {
    int d0 = 2 * p;
    int h  = d0 >> 7;             // E_ = 128
    int e0 = d0 & (E_ - 1);
    int pr = e0 >> 1;
    float c = cosT[n * (E_ / 2) + pr];
    float s = sinT[n * (E_ / 2) + pr];
    float q0 = row[d0]     * rq * nqw[d0];
    float q1 = row[d0 + 1] * rq * nqw[d0 + 1];
    float k0 = row[D_ + d0]     * rk * nkw[d0];
    float k1 = row[D_ + d0 + 1] * rk * nkw[d0 + 1];
    size_t o = (((size_t)(b * H_ + h)) * N_ + n) * E_ + e0;   // even -> 4B aligned
    *(unsigned int*)(qb + o) =
        (unsigned)f32_to_bf16(q0 * c - q1 * s) | ((unsigned)f32_to_bf16(q0 * s + q1 * c) << 16);
    *(unsigned int*)(kb + o) =
        (unsigned)f32_to_bf16(k0 * c - k1 * s) | ((unsigned)f32_to_bf16(k0 * s + k1 * c) << 16);
    *(unsigned int*)(vb + o) =
        (unsigned)f32_to_bf16(row[2 * D_ + d0]) | ((unsigned)f32_to_bf16(row[2 * D_ + d0 + 1]) << 16);
  }
}

// ---------------------------------------------------------------------------
// Flash attention per (b,h). Block = 128 threads (4 waves), 64 q rows/block,
// each wave owns a 16-row Q tile resident in registers (4 A-frags, E=128).
// K tile [64][128] in LDS via async global->LDS copy (ASYNCcnt) when available;
// V tile transposed [128][64] via VGPR path. Online softmax via width-16
// shuffles; P reshaped C->A layout through wave-private LDS staging.
// ---------------------------------------------------------------------------
__global__ __launch_bounds__(128) void flash_attn(
    const unsigned short* __restrict__ Qa,
    const unsigned short* __restrict__ Ka,
    const unsigned short* __restrict__ Va,
    float* __restrict__ O)
{
  __shared__ __align__(16) unsigned short Ks[64 * 128];   // [key][e]
  __shared__ __align__(16) unsigned short Vt[128 * 64];   // [e][key]
  __shared__ __align__(16) unsigned short Ps[4][16 * 32]; // per-wave P staging

  const int bh = blockIdx.y;
  const int b  = bh >> 4;          // H_ = 16
  const int h  = bh & (H_ - 1);
  const int q0 = blockIdx.x * 64;
  const int t    = threadIdx.x;
  const int wave = t >> 5;
  const int lane = t & 31;
  const int lh   = lane >> 4;
  const int l15  = lane & 15;

  const unsigned short* Qb = Qa + (size_t)bh * N_ * E_;
  const unsigned short* Kb = Ka + (size_t)bh * N_ * E_;
  const unsigned short* Vb = Va + (size_t)bh * N_ * E_;

  // Q fragments: row = l15 (within wave's 16-row tile), 4 k-chunks of 32
  Frag16 qf[4];
  {
    const u32x4* pq = (const u32x4*)(Qb + (size_t)(q0 + wave * 16 + l15) * E_);
#pragma unroll
    for (int s = 0; s < 4; ++s) {
      qf[s].q[0] = pq[s * 4 + 2 * lh];
      qf[s].q[1] = pq[s * 4 + 2 * lh + 1];
    }
  }

  f32x8 Of[8];
#pragma unroll
  for (int c = 0; c < 8; ++c)
#pragma unroll
    for (int r = 0; r < 8; ++r) Of[c][r] = 0.f;
  float mrow[8], lrow[8];
#pragma unroll
  for (int r = 0; r < 8; ++r) { mrow[r] = -1e30f; lrow[r] = 0.f; }

  const float scale = 0.08838834764831845f;   // 1/sqrt(128)

  for (int j0 = 0; j0 < N_; j0 += 64) {
    // ---- K tile [64][128] ----
#if USE_ASYNC_LDS
    {
      const unsigned short* gsrc = Kb + (size_t)j0 * E_;
#pragma unroll
      for (int i = 0; i < 8; ++i) {
        int idx = t + i * 128;   // 1024 x b128 transfers
        AS1 i32x4* g = (AS1 i32x4*)(unsigned long long)(size_t)(gsrc + idx * 8);
        AS3 i32x4* l = (AS3 i32x4*)(unsigned int)(size_t)(Ks + idx * 8);
        __builtin_amdgcn_global_load_async_to_lds_b128(g, l, 0, 0);
      }
    }
#else
    {
      const unsigned int* Kg   = (const unsigned int*)(Kb + (size_t)j0 * E_);
      unsigned int*       Ks32 = (unsigned int*)Ks;
#pragma unroll
      for (int i = 0; i < 32; ++i) Ks32[t + i * 128] = Kg[t + i * 128];
    }
#endif
    // ---- V tile transposed ----
#pragma unroll
    for (int i = 0; i < 64; ++i) {
      int idx = t + i * 128;
      int key = idx >> 7, e = idx & (E_ - 1);
      Vt[e * 64 + key] = Vb[(size_t)(j0 + key) * E_ + e];
    }
#if USE_ASYNC_LDS
    __builtin_amdgcn_s_wait_asynccnt(0);
#endif
    __syncthreads();

#pragma unroll
    for (int sub = 0; sub < 2; ++sub) {          // 32 keys per step
      f32x8 S0, S1;
#pragma unroll
      for (int r = 0; r < 8; ++r) { S0[r] = 0.f; S1[r] = 0.f; }
#pragma unroll
      for (int s = 0; s < 4; ++s) {
        Frag16 kb0, kb1;     // B-frag: col = key-in-16-block, K = feature chunk
        const u32x4* p0 = (const u32x4*)(Ks + (sub * 32 + l15) * 128 + s * 32 + lh * 16);
        kb0.q[0] = p0[0]; kb0.q[1] = p0[1];
        const u32x4* p1 = (const u32x4*)(Ks + (sub * 32 + 16 + l15) * 128 + s * 32 + lh * 16);
        kb1.q[0] = p1[0]; kb1.q[1] = p1[1];
        S0 = wmma_bf16(qf[s], kb0, S0);
        S1 = wmma_bf16(qf[s], kb1, S1);
      }

      // online softmax: row r lives in (elem r, lane-half lh); reduce over 16 lanes
      float p0v[8], p1v[8], alpha[8];
#pragma unroll
      for (int r = 0; r < 8; ++r) {
        float s0 = S0[r] * scale, s1 = S1[r] * scale;
        float vmax = fmaxf(s0, s1);
#pragma unroll
        for (int o = 1; o < 16; o <<= 1) vmax = fmaxf(vmax, __shfl_xor(vmax, o, 16));
        float nm = fmaxf(mrow[r], vmax);
        float al = __expf(mrow[r] - nm);
        float e0 = __expf(s0 - nm), e1 = __expf(s1 - nm);
        float rs = e0 + e1;
#pragma unroll
        for (int o = 1; o < 16; o <<= 1) rs += __shfl_xor(rs, o, 16);
        lrow[r] = lrow[r] * al + rs;
        mrow[r] = nm;
        alpha[r] = al; p0v[r] = e0; p1v[r] = e1;
      }
#pragma unroll
      for (int c = 0; c < 8; ++c)
#pragma unroll
        for (int r = 0; r < 8; ++r) Of[c][r] *= alpha[r];

      // stage P (C layout) -> wave-private LDS -> reload as A fragment
#pragma unroll
      for (int r = 0; r < 8; ++r) {
        int mr = lh * 8 + r;
        Ps[wave][mr * 32 + l15]      = f32_to_bf16(p0v[r]);
        Ps[wave][mr * 32 + 16 + l15] = f32_to_bf16(p1v[r]);
      }
      Frag16 pf;                      // same-wave DS ops are in-order
      const u32x4* pp = (const u32x4*)(&Ps[wave][l15 * 32 + lh * 16]);
      pf.q[0] = pp[0]; pf.q[1] = pp[1];

#pragma unroll
      for (int c = 0; c < 8; ++c) {   // O[16x128] += P[16x32] @ V[32x128]
        Frag16 vf;
        const u32x4* pv = (const u32x4*)(Vt + (c * 16 + l15) * 64 + sub * 32 + lh * 16);
        vf.q[0] = pv[0]; vf.q[1] = pv[1];
        Of[c] = wmma_bf16(pf, vf, Of[c]);
      }
    }
    __syncthreads();
  }

#pragma unroll
  for (int r = 0; r < 8; ++r) {
    float inv = 1.f / lrow[r];
    int n = q0 + wave * 16 + lh * 8 + r;
    float* dst = O + ((size_t)b * N_ + n) * D_ + h * E_;
#pragma unroll
    for (int c = 0; c < 8; ++c) dst[c * 16 + l15] = Of[c][r] * inv;
  }
}

// ---------------------------------------------------------------------------
extern "C" void kernel_launch(void* const* d_in, const int* in_sizes, int n_in,
                              void* d_out, int out_size, void* d_ws, size_t ws_size,
                              hipStream_t stream)
{
  (void)in_sizes; (void)n_in; (void)out_size; (void)ws_size;
  const float* spatial  = (const float*)d_in[0];
  const float* w_qkv    = (const float*)d_in[1];
  const float* b_qkv    = (const float*)d_in[2];
  const float* nqw      = (const float*)d_in[3];
  const float* nkw      = (const float*)d_in[4];
  const float* w_out    = (const float*)d_in[5];
  const float* b_out    = (const float*)d_in[6];
  const float* rope_cos = (const float*)d_in[7];
  const float* rope_sin = (const float*)d_in[8];
  const float* residual = (const float*)d_in[9];
  const float* gate     = (const float*)d_in[10];

  // Workspace layout (~176 MB):
  //   qkv fp32 [4096,6144] | q,k,v bf16 [B,H,N,E] | attn fp32 [B,N,D]
  char* ws = (char*)d_ws;
  float* qkv = (float*)ws;
  size_t off = (size_t)4096 * 6144 * sizeof(float);              // 96 MB
  unsigned short* qb = (unsigned short*)(ws + off); off += (size_t)B_ * D_ * N_ * 2;
  unsigned short* kb = (unsigned short*)(ws + off); off += (size_t)B_ * D_ * N_ * 2;
  unsigned short* vb = (unsigned short*)(ws + off); off += (size_t)B_ * D_ * N_ * 2;
  float* attn = (float*)(ws + off);                              // 32 MB

  // 1) QKV projection: [4096,2048] @ [2048,6144] + bias
  dim3 g1(6144 / 64, 4096 / 128);
  gemm_bf16_epilogue<<<g1, 128, 0, stream>>>(spatial, w_qkv, b_qkv, qkv,
                                             nullptr, nullptr,
                                             D_, 3 * D_, N_, 0);
  // 2) RMSNorm + RoPE + head split + bf16 pack
  norm_rope_pack<<<B_ * N_, 256, 0, stream>>>(qkv, nqw, nkw, rope_cos, rope_sin,
                                              qb, kb, vb);
  // 3) Flash SDPA per (b,h)
  dim3 g3(N_ / 64, B_ * H_);
  flash_attn<<<g3, 128, 0, stream>>>(qb, kb, vb, attn);
  // 4) Out projection + residual + gate
  dim3 g4(D_ / 64, 4096 / 128);
  gemm_bf16_epilogue<<<g4, 128, 0, stream>>>(attn, w_out, b_out, (float*)d_out,
                                             residual, gate,
                                             D_, D_, N_, 1);
}
